// GraphormerAttentionLayer_56341380989043
// MI455X (gfx1250) — matrix operand order repaired and checked by reading, not compiled, not run on hardware
//
#include <hip/hip_runtime.h>
#include <hip/hip_bf16.h>

typedef _Float16 v16h __attribute__((ext_vector_type(16)));
typedef float    v8f  __attribute__((ext_vector_type(8)));

__device__ __forceinline__ v8f wmma_f16(v16h a, v16h b, v8f c) {
    return __builtin_amdgcn_wmma_f32_16x16x32_f16(false, a, false, b, (short)0, c, false, false);
}

// A-fragment k index (contiguous order, matches B): half h of lane-group g -> k = 16g + h
__device__ __forceinline__ int a_kof(int h, int g) {
    return g * 16 + h;
}

// DPP16 all-reduce max over a 16-lane row (pure VALU, no LDS traffic).
template<int CTRL>
__device__ __forceinline__ float dpp_max_step(float x) {
    int r = __builtin_amdgcn_mov_dpp(__builtin_bit_cast(int, x), CTRL, 0xF, 0xF, true);
    return fmaxf(x, __builtin_bit_cast(float, r));
}
__device__ __forceinline__ float row16_allmax(float x) {
    x = dpp_max_step<0x128>(x);   // row_ror:8
    x = dpp_max_step<0x124>(x);   // row_ror:4
    x = dpp_max_step<0x4E>(x);    // quad_perm [2,3,0,1]  (xor 2)
    x = dpp_max_step<0xB1>(x);    // quad_perm [1,0,3,2]  (xor 1)
    return x;
}

// ---------------------------------------------------------------------------
// Generic 512-K GEMM: Y[M,512] = A[M,512] @ W[512,512]^T (+bias)(*scale)
// Block: 256 threads (8 waves), tile 128x64, each wave a 32x32 (2x2 WMMA) tile.
// OUT_MODE: 0 = f32 row-major [M,512]
//           1 = f16 scatter [B,H,N,32]   (Q/K)
//           2 = f16 scatter [B,H,32,N]   (V transposed for B-fragment loads)
// ---------------------------------------------------------------------------
constexpr int DM  = 512;
constexpr int BM  = 128;
constexpr int BN  = 64;
constexpr int BK  = 32;
constexpr int LDT = 48;   // LDS row stride (halfs)

template<bool A_F16, int OUT_MODE>
__global__ __launch_bounds__(256) void gemm512(
    const float* __restrict__ Af32, const _Float16* __restrict__ Af16,
    const float* __restrict__ W,    const float* __restrict__ bias,
    _Float16* __restrict__ out16,   float* __restrict__ out32, float scale)
{
    __shared__ _Float16 Xs[BM * LDT];
    __shared__ _Float16 Ws[BN * LDT];

    const int tid  = threadIdx.x;
    const int lane = tid & 31;
    const int wave = tid >> 5;
    const int m0   = blockIdx.x * BM;
    const int n0   = blockIdx.y * BN;
    const int wm   = (wave >> 1) * 32;   // 0,32,64,96
    const int wn   = (wave & 1) * 32;    // 0,32
    const int lm   = lane & 15;
    const int g    = lane >> 4;

    v8f c[2][2] = {};

    for (int k0 = 0; k0 < DM; k0 += BK) {
        {   // stage A tile (128x32): 2 threads per row, 16 elems each
            const int r  = tid >> 1;
            const int kp = (tid & 1) * 16;
            if (A_F16) {
                const _Float16* src = Af16 + (size_t)(m0 + r) * DM + k0 + kp;
                #pragma unroll
                for (int i = 0; i < 16; ++i) Xs[r * LDT + kp + i] = src[i];
            } else {
                const float* src = Af32 + (size_t)(m0 + r) * DM + k0 + kp;
                #pragma unroll
                for (int i = 0; i < 16; ++i) Xs[r * LDT + kp + i] = (_Float16)src[i];
            }
        }
        {   // stage W tile (64x32), row n holds k contiguous
            const int nr = tid >> 2;
            const int kw = (tid & 3) * 8;
            const float* src = W + (size_t)(n0 + nr) * DM + k0 + kw;
            #pragma unroll
            for (int i = 0; i < 8; ++i) Ws[nr * LDT + kw + i] = (_Float16)src[i];
        }
        __syncthreads();

        v16h a[2], b[2];
        #pragma unroll
        for (int h = 0; h < 16; ++h) {
            const int ka = a_kof(h, g);
            a[0][h] = Xs[(wm + lm) * LDT + ka];
            a[1][h] = Xs[(wm + 16 + lm) * LDT + ka];
            const int kb = h + 16 * g;           // B-fragment k index
            b[0][h] = Ws[(wn + lm) * LDT + kb];
            b[1][h] = Ws[(wn + 16 + lm) * LDT + kb];
        }
        #pragma unroll
        for (int mt = 0; mt < 2; ++mt)
            #pragma unroll
            for (int nt = 0; nt < 2; ++nt)
                c[mt][nt] = wmma_f16(a[mt], b[nt], c[mt][nt]);
        __syncthreads();
    }

    #pragma unroll
    for (int mt = 0; mt < 2; ++mt) {
        #pragma unroll
        for (int nt = 0; nt < 2; ++nt) {
            #pragma unroll
            for (int v = 0; v < 8; ++v) {
                const int mg = m0 + wm + mt * 16 + v + 8 * g;
                const int ng = n0 + wn + nt * 16 + lm;
                float val = (c[mt][nt][v] + bias[ng]) * scale;
                if (OUT_MODE == 0) {
                    out32[(size_t)mg * DM + ng] = val;
                } else {
                    const int hh = ng >> 5, dk = ng & 31;
                    const int bb = mg >> 10, ns = mg & 1023;
                    if (OUT_MODE == 1)
                        out16[(((size_t)(bb * 16 + hh) * 1024) + ns) * 32 + dk] = (_Float16)val;
                    else  // V transposed: [B,H,32,N]
                        out16[(((size_t)(bb * 16 + hh) * 32) + dk) * 1024 + ns] = (_Float16)val;
                }
            }
        }
    }
}

// ---------------------------------------------------------------------------
// Flash attention: block = 512 threads = 16 waves = all 16 heads of one
// (batch b, 16-row Q tile). attn_bias [16 x 32 x 16h] chunks staged to LDS
// with coalesced loads, shared by all heads. Softmax row-max by DPP16
// butterfly (pure VALU); row-sum by an extra WMMA against an all-ones matrix.
// ---------------------------------------------------------------------------
__global__ __launch_bounds__(512) void attn_kernel(
    const _Float16* __restrict__ Q, const _Float16* __restrict__ K,
    const _Float16* __restrict__ Vt, const float* __restrict__ bias,
    _Float16* __restrict__ O)
{
    __shared__ float    Blds[16 * 32 * 16];     // [i][j][h] f32, 32 KB
    __shared__ _Float16 Plds[16][16 * 36];      // per-wave P transpose buffer

    const int tid  = threadIdx.x;
    const int lane = tid & 31;
    const int wave = tid >> 5;                  // head index 0..15
    const int b    = blockIdx.x >> 6;           // batch
    const int qt   = blockIdx.x & 63;           // q tile
    const int h    = wave;
    const int lm   = lane & 15;
    const int g    = lane >> 4;
    const int bh   = b * 16 + h;
    const size_t seq = (size_t)bh * 1024;

    // Q A-fragment (rows qt*16..+15, k = d 0..31)
    v16h aQ;
    {
        const _Float16* qrow = Q + (seq + qt * 16 + lm) * 32;
        #pragma unroll
        for (int hh = 0; hh < 16; ++hh) aQ[hh] = qrow[a_kof(hh, g)];
    }

    v16h ones;
    #pragma unroll
    for (int hh = 0; hh < 16; ++hh) ones[hh] = (_Float16)1.0f;

    v8f o0 = {}, o1 = {};
    float mrow[8], lrow[8], alpha[8];
    #pragma unroll
    for (int v = 0; v < 8; ++v) { mrow[v] = -1e30f; lrow[v] = 0.0f; }

    _Float16* P = &Plds[wave][0];

    for (int jb = 0; jb < 64; ++jb) {
        const int j0 = jb * 32;

        // ---- cooperative coalesced bias stage ----
        __syncthreads();   // previous iteration finished reading Blds
        {
            const int bi  = tid >> 5;            // 0..15
            const int off = (tid & 31) * 16;     // 0..496
            const float4* src = (const float4*)(bias +
                (((size_t)(qt * 16 + bi) * 1024 + j0) * 16) + off);
            float4* dst = (float4*)(Blds + bi * 512 + off);
            #pragma unroll
            for (int i = 0; i < 4; ++i) dst[i] = src[i];
        }
        __syncthreads();   // bias tile ready

        // ---- S = Q K^T : contiguous 32B B-fragment loads ----
        v16h bk0 = *(const v16h*)(K + (seq + j0 + lm) * 32 + g * 16);
        v16h bk1 = *(const v16h*)(K + (seq + j0 + 16 + lm) * 32 + g * 16);

        v8f z = {};
        v8f s0 = wmma_f16(aQ, bk0, z);
        v8f s1 = wmma_f16(aQ, bk1, z);

        #pragma unroll
        for (int v = 0; v < 8; ++v) {
            const int il = v + 8 * g;                       // local query row
            s0[v] += Blds[(il * 32 + lm) * 16 + h];
            s1[v] += Blds[(il * 32 + 16 + lm) * 16 + h];

            const float mx    = row16_allmax(fmaxf(s0[v], s1[v]));
            const float mnew  = fmaxf(mrow[v], mx);
            alpha[v] = __expf(mrow[v] - mnew);
            mrow[v] = mnew;
            o0[v] *= alpha[v];
            o1[v] *= alpha[v];
            const float p0 = __expf(s0[v] - mnew);
            const float p1 = __expf(s1[v] - mnew);
            // C-layout -> LDS row-major P[16][32]
            P[il * 36 + lm]      = (_Float16)p0;
            P[il * 36 + 16 + lm] = (_Float16)p1;
        }
        asm volatile("s_wait_dscnt 0" ::: "memory");

        // P A-fragment (transpose through LDS)
        v16h aP;
        #pragma unroll
        for (int hh = 0; hh < 16; ++hh) aP[hh] = P[lm * 36 + a_kof(hh, g)];

        // row-sum of P via the matrix engine: rs(m,n) = sum_k P(m,k)
        v8f rs = wmma_f16(aP, ones, z);
        #pragma unroll
        for (int v = 0; v < 8; ++v) lrow[v] = lrow[v] * alpha[v] + rs[v];

        // V B-fragments from transposed V: contiguous 32B per lane
        v16h bv0 = *(const v16h*)(Vt + ((size_t)bh * 32 + lm) * 1024 + j0 + g * 16);
        v16h bv1 = *(const v16h*)(Vt + ((size_t)bh * 32 + 16 + lm) * 1024 + j0 + g * 16);

        o0 = wmma_f16(aP, bv0, o0);
        o1 = wmma_f16(aP, bv1, o1);
    }

    // normalize and store O as f16 [B,N, H*32] = [8192, 512]
    #pragma unroll
    for (int v = 0; v < 8; ++v) {
        const float inv = 1.0f / lrow[v];
        const int ig = qt * 16 + v + 8 * g;
        const size_t orow = ((size_t)b * 1024 + ig) * 512 + h * 32;
        O[orow + lm]      = (_Float16)(o0[v] * inv);
        O[orow + 16 + lm] = (_Float16)(o1[v] * inv);
    }
}

// ---------------------------------------------------------------------------
extern "C" void kernel_launch(void* const* d_in, const int* in_sizes, int n_in,
                              void* d_out, int out_size, void* d_ws, size_t ws_size,
                              hipStream_t stream) {
    const float* x  = (const float*)d_in[0];
    const float* ab = (const float*)d_in[1];
    const float* wq = (const float*)d_in[2];
    const float* bq = (const float*)d_in[3];
    const float* wk = (const float*)d_in[4];
    const float* bk = (const float*)d_in[5];
    const float* wv = (const float*)d_in[6];
    const float* bv = (const float*)d_in[7];
    const float* wo = (const float*)d_in[8];
    const float* bo = (const float*)d_in[9];
    float* out = (float*)d_out;

    const size_t MAT = (size_t)8192 * 512;   // halfs per buffer
    _Float16* Qb = (_Float16*)d_ws;
    _Float16* Kb = Qb + MAT;
    _Float16* Vb = Kb + MAT;   // stored transposed [B,H,32,N]
    _Float16* Ob = Vb + MAT;

    dim3 grid(8192 / BM, DM / BN);           // 64 x 8
    const float qscale = 0.17677669529663687f;   // 1/sqrt(32)

    gemm512<false, 1><<<grid, 256, 0, stream>>>(x, nullptr, wq, bq, Qb, nullptr, qscale);
    gemm512<false, 1><<<grid, 256, 0, stream>>>(x, nullptr, wk, bk, Kb, nullptr, 1.0f);
    gemm512<false, 2><<<grid, 256, 0, stream>>>(x, nullptr, wv, bv, Vb, nullptr, 1.0f);

    attn_kernel<<<512, 512, 0, stream>>>(Qb, Kb, Vb, ab, Ob);

    gemm512<true, 0><<<grid, 256, 0, stream>>>(nullptr, Ob, wo, bo, nullptr, out, 1.0f);
}
